// GNNlayer_16707422781845
// MI455X (gfx1250) — matrix operand order repaired and checked by reading, not compiled, not run on hardware
//
#include <hip/hip_runtime.h>
#include <hip/hip_bf16.h>

#define DFEAT 128

typedef __attribute__((ext_vector_type(16))) __bf16 v16bf;
typedef __attribute__((ext_vector_type(8)))  float  v8f;

// ---------------------------------------------------------------------------
// Kernel 1: zero the output accumulator (float4 stores, coalesced)
// ---------------------------------------------------------------------------
__global__ __launch_bounds__(256) void zero_out_kernel(float4* __restrict__ out,
                                                       int n_vec4) {
    int i = blockIdx.x * 256 + threadIdx.x;
    if (i < n_vec4) {
        out[i] = make_float4(0.f, 0.f, 0.f, 0.f);
    }
}

// ---------------------------------------------------------------------------
// Kernel 2: h = feat @ W^T + b via v_wmma_f32_16x16x32_bf16.
// fp32 inputs are split hi/lo into bf16 (x = hi + lo, lo = bf16(x - f32(hi)))
// and three WMMAs per K-step recover ~fp32 accuracy:
//   fh*wh + fh*wl + fl*wh   (fl*wl term is ~2^-16 relative, dropped)
//
// One wave (32 lanes) computes one 16x16 output tile; K=128 in 4 steps of 32.
// 256-thread blocks = 8 waves = 8 tiles per block.
//
// A-matrix (16x32 bf16, ISA 7.12.2): lane L holds row M=L%16;
//   VGPR v<4 : K = 2v     + 8*(L/16)   (pair K, K+1)
//   VGPR v>=4: K = 16+2(v-4) + 8*(L/16)
// B-matrix (32x16 bf16): lane L holds col N=L%16;
//   VGPR v: same K mapping as A         (pair K, K+1)
// C/D (16x16 f32): VGPR r, lane L -> (M = r + 8*(L/16), N = L%16)
// ---------------------------------------------------------------------------
__global__ __launch_bounds__(256) void gemm_wmma_kernel(
        const float* __restrict__ feat,   // [n_nodes, 128] row-major
        const float* __restrict__ W,      // [128, 128] row-major; h=feat@W^T
        const float* __restrict__ bias,   // [128]
        float* __restrict__ h,            // [n_nodes, 128]
        int n_tiles)                      // (n_nodes/16)*8
{
    const int lane = threadIdx.x & 31;
    const int wave = threadIdx.x >> 5;
    const int tile = blockIdx.x * 8 + wave;
    if (tile >= n_tiles) return;          // whole wave exits together

    const int col  = (tile & 7) << 4;     // 8 column tiles of 16 over D=128
    const int row  = (tile >> 3) << 4;    // 16-row tile
    const int m    = lane & 15;
    const int half = lane >> 4;           // 0: lanes 0-15, 1: lanes 16-31

    // init accumulator with bias: C[M][N] = bias[col + N], N = lane%16
    v8f acc;
    {
        const float bv = bias[col + m];
        #pragma unroll
        for (int r = 0; r < 8; ++r) acc[r] = bv;
    }

    const float* __restrict__ arow = feat + (size_t)(row + m) * DFEAT;
    const float* __restrict__ wrow = W    + (size_t)(col + m) * DFEAT;

    for (int k0 = 0; k0 < DFEAT; k0 += 32) {
        v16bf a_hi, a_lo, b_hi, b_lo;

        // ---- A tile: feat[row+m][k0 + Kv .. +1], hi/lo split ----
        #pragma unroll
        for (int v = 0; v < 8; ++v) {
            const int k = ((v < 4) ? (2 * v) : (16 + 2 * (v - 4))) + 8 * half + k0;
            const float x0 = arow[k];
            const float x1 = arow[k + 1];
            const __bf16 h0 = (__bf16)x0;
            const __bf16 h1 = (__bf16)x1;
            a_hi[2 * v]     = h0;
            a_hi[2 * v + 1] = h1;
            a_lo[2 * v]     = (__bf16)(x0 - (float)h0);
            a_lo[2 * v + 1] = (__bf16)(x1 - (float)h1);
        }

        // ---- B tile: W[col+n][k0 + Kv .. +1], n = lane%16, hi/lo split ----
        #pragma unroll
        for (int v = 0; v < 8; ++v) {
            const int k = ((v < 4) ? (2 * v) : (16 + 2 * (v - 4))) + 8 * half + k0;
            const float x0 = wrow[k];
            const float x1 = wrow[k + 1];
            const __bf16 h0 = (__bf16)x0;
            const __bf16 h1 = (__bf16)x1;
            b_hi[2 * v]     = h0;
            b_hi[2 * v + 1] = h1;
            b_lo[2 * v]     = (__bf16)(x0 - (float)h0);
            b_lo[2 * v + 1] = (__bf16)(x1 - (float)h1);
        }

        // 3-product bf16 emulation of fp32 GEMM on the matrix pipe
        acc = __builtin_amdgcn_wmma_f32_16x16x32_bf16(
                  false, a_hi, false, b_hi, (short)0, acc, false, false);
        acc = __builtin_amdgcn_wmma_f32_16x16x32_bf16(
                  false, a_hi, false, b_lo, (short)0, acc, false, false);
        acc = __builtin_amdgcn_wmma_f32_16x16x32_bf16(
                  false, a_lo, false, b_hi, (short)0, acc, false, false);
    }

    // store D: VGPR r -> row (row + r + 8*half), col (col + m); 64B contiguous
    #pragma unroll
    for (int r = 0; r < 8; ++r) {
        h[(size_t)(row + r + 8 * half) * DFEAT + col + m] = acc[r];
    }
}

// ---------------------------------------------------------------------------
// Kernel 3: out[dst[e]] += h[src[e]]   (segment-sum over 3.2M edges)
// One wave per edge: all 32 lanes broadcast-read src/dst, each lane float4-
// loads 16B of the 512B source row (fully coalesced) and issues 4 no-return
// global_atomic_add_f32. h + out + edge lists fit in the 192MB L2, so this is
// L2-resident atomic traffic, not HBM traffic.
// ---------------------------------------------------------------------------
__global__ __launch_bounds__(256) void scatter_add_kernel(
        const float* __restrict__ h,
        const int* __restrict__ src,
        const int* __restrict__ dst,
        float* __restrict__ out,
        int n_edges)
{
    const int t = blockIdx.x * 256 + threadIdx.x;   // wave-aligned: t/32 const per wave
    const int e = t >> 5;
    if (e >= n_edges) return;
    const int chunk = t & 31;                       // 32 chunks x float4 = 128 floats

    const int s = src[e];
    const int d = dst[e];

    const float4 v = *(const float4*)(h + (size_t)s * DFEAT + chunk * 4);
    float* o = out + (size_t)d * DFEAT + chunk * 4;

    unsafeAtomicAdd(o + 0, v.x);   // lowers to global_atomic_add_f32 (no return)
    unsafeAtomicAdd(o + 1, v.y);
    unsafeAtomicAdd(o + 2, v.z);
    unsafeAtomicAdd(o + 3, v.w);
}

// ---------------------------------------------------------------------------
// Host launch
// ---------------------------------------------------------------------------
extern "C" void kernel_launch(void* const* d_in, const int* in_sizes, int n_in,
                              void* d_out, int out_size, void* d_ws, size_t ws_size,
                              hipStream_t stream) {
    const float* feat = (const float*)d_in[0];   // [n_nodes, 128]
    const float* W    = (const float*)d_in[1];   // [128, 128]
    const float* bias = (const float*)d_in[2];   // [128]
    const int*   src  = (const int*)d_in[3];     // [n_edges]
    const int*   dst  = (const int*)d_in[4];     // [n_edges]
    float*       out  = (float*)d_out;           // [n_nodes, 128]
    float*       h    = (float*)d_ws;            // [n_nodes, 128] scratch

    const int n_nodes = in_sizes[0] / DFEAT;
    const int n_edges = in_sizes[3];

    // 1) zero output accumulator
    {
        const int n_vec4 = (n_nodes * DFEAT) / 4;
        const int blocks = (n_vec4 + 255) / 256;
        zero_out_kernel<<<blocks, 256, 0, stream>>>((float4*)out, n_vec4);
    }

    // 2) h = feat @ W^T + b  (WMMA, one wave per 16x16 tile)
    {
        const int n_tiles = (n_nodes / 16) * (DFEAT / 16);  // 6250 * 8 = 50000
        const int blocks  = (n_tiles + 7) / 8;              // 8 waves per block
        gemm_wmma_kernel<<<blocks, 256, 0, stream>>>(feat, W, bias, h, n_tiles);
    }

    // 3) scatter: one wave per edge
    {
        const long long total  = (long long)n_edges * 32;
        const int       blocks = (int)((total + 255) / 256);
        scatter_add_kernel<<<blocks, 256, 0, stream>>>(h, src, dst, out, n_edges);
    }
}
